// Attention_neo_51264729645449
// MI455X (gfx1250) — compile-verified, matrix operand order checked by reading, not run on hardware
//
#include <hip/hip_runtime.h>
#include <hip/hip_bf16.h>

// ---- CDNA5 wave32 WMMA attention ----
// B=4, S=2048, D=1024, H=16, dh=64.  All GEMMs use v_wmma_f32_16x16x32_bf16.

typedef __attribute__((ext_vector_type(16))) __bf16          v16bf;
typedef __attribute__((ext_vector_type(8)))  float           v8f;
typedef __attribute__((ext_vector_type(8)))  unsigned short  v8us;
typedef __attribute__((ext_vector_type(16))) unsigned short  v16us;

#define BB 4
#define SS 2048
#define DD 1024
#define HH 16
#define DH 64
#define MM (BB * SS)   // 8192

// fp32 -> bf16 (round to nearest even)
__device__ __forceinline__ unsigned short f2bf(float f) {
  unsigned int u = __builtin_bit_cast(unsigned int, f);
  unsigned int r = u + 0x7FFFu + ((u >> 16) & 1u);
  return (unsigned short)(r >> 16);
}

// Load a 16x32 bf16 fragment (A-style; B 32x16 uses the same lane pattern with
// lane = N) from row-major LDS [rows][stride] at (rowBase, colBase..colBase+31).
__device__ __forceinline__ v16bf ld_frag(const unsigned short* base, int stride,
                                         int rowBase, int colBase) {
  int lane = threadIdx.x & 31;
  int r = rowBase + (lane & 15);
  int kb = (lane & 16) ? 8 : 0;
  const unsigned short* p = base + r * stride + colBase + kb;
  v8us lo = *reinterpret_cast<const v8us*>(p);         // ds_load_b128
  v8us hi = *reinterpret_cast<const v8us*>(p + 16);    // ds_load_b128
  v16us u;
#pragma unroll
  for (int i = 0; i < 8; ++i) { u[i] = lo[i]; u[i + 8] = hi[i]; }
  return __builtin_bit_cast(v16bf, u);
}

__device__ __forceinline__ v8f wmma_bf16(v16bf a, v16bf b, v8f c) {
  return __builtin_amdgcn_wmma_f32_16x16x32_bf16(false, a, false, b, (short)0, c,
                                                 false, false);
}

// ---------------------------------------------------------------------------
// C[M,N] = A[M,K] * W[N,K]^T (+bias).  M=8192, N=K=1024.
// Block: 256 thr (8 waves), tile 128x128, K-step 64 (16 WMMAs per barrier pair).
// Wave grid 4(M) x 2(N), each wave: 32x64 -> 2x4 accumulators of 16x16.
// PERMUTE: scatter into [B,H,S,dh] (head-split) instead of [M,N].
// ---------------------------------------------------------------------------
template <bool PERMUTE, bool BIAS>
__global__ __launch_bounds__(256) void linear_wmma(const float* __restrict__ A,
                                                   const float* __restrict__ W,
                                                   const float* __restrict__ bias,
                                                   float* __restrict__ C) {
  __shared__ unsigned short sA[128 * 72];
  __shared__ unsigned short sB[128 * 72];

  const int tid  = threadIdx.x;
  const int lane = tid & 31;
  const int wave = tid >> 5;
  const int mw   = wave >> 1;   // 0..3
  const int nw   = wave & 1;    // 0..1
  const int mbase = blockIdx.y * 128;
  const int nbase = blockIdx.x * 128;

  v8f acc[2][4];
#pragma unroll
  for (int i = 0; i < 2; ++i)
#pragma unroll
    for (int j = 0; j < 4; ++j) acc[i][j] = v8f{0, 0, 0, 0, 0, 0, 0, 0};

  const int row0 = tid >> 4;        // 0..15
  const int c4   = (tid & 15) * 4;  // 0..60

  for (int kt = 0; kt < 16; ++kt) {
    const int kcol = kt * 64 + c4;
#pragma unroll
    for (int i = 0; i < 8; ++i) {
      const int r = row0 + i * 16;
      float4 av = *reinterpret_cast<const float4*>(A + (size_t)(mbase + r) * DD + kcol);
      float4 wv = *reinterpret_cast<const float4*>(W + (size_t)(nbase + r) * DD + kcol);
      unsigned int* pa = reinterpret_cast<unsigned int*>(&sA[r * 72 + c4]);
      unsigned int* pb = reinterpret_cast<unsigned int*>(&sB[r * 72 + c4]);
      pa[0] = (unsigned int)f2bf(av.x) | ((unsigned int)f2bf(av.y) << 16);
      pa[1] = (unsigned int)f2bf(av.z) | ((unsigned int)f2bf(av.w) << 16);
      pb[0] = (unsigned int)f2bf(wv.x) | ((unsigned int)f2bf(wv.y) << 16);
      pb[1] = (unsigned int)f2bf(wv.z) | ((unsigned int)f2bf(wv.w) << 16);
    }
    __syncthreads();

#pragma unroll
    for (int ks = 0; ks < 2; ++ks) {
      v16bf af0 = ld_frag(sA, 72, mw * 32, ks * 32);
      v16bf af1 = ld_frag(sA, 72, mw * 32 + 16, ks * 32);
#pragma unroll
      for (int ni = 0; ni < 4; ++ni) {
        v16bf bf = ld_frag(sB, 72, nw * 64 + ni * 16, ks * 32);
        acc[0][ni] = wmma_bf16(af0, bf, acc[0][ni]);
        acc[1][ni] = wmma_bf16(af1, bf, acc[1][ni]);
      }
    }
    __syncthreads();
  }

  const int lg = lane >> 4;
  const int ln = lane & 15;
#pragma unroll
  for (int mi = 0; mi < 2; ++mi) {
#pragma unroll
    for (int ni = 0; ni < 4; ++ni) {
      const int n = nbase + nw * 64 + ni * 16 + ln;
      float bv = 0.0f;
      if (BIAS) bv = bias[n];
#pragma unroll
      for (int v = 0; v < 8; ++v) {
        const int m = mbase + mw * 32 + mi * 16 + v + 8 * lg;
        const float val = acc[mi][ni][v] + bv;
        if (PERMUTE) {
          const int b = m >> 11, s = m & 2047;
          const int h = n >> 6,  d = n & 63;
          C[(((size_t)(b * HH + h) * SS) + s) * DH + d] = val;
        } else {
          C[(size_t)m * DD + n] = val;
        }
      }
    }
  }
}

// ---------------------------------------------------------------------------
// Flash attention.  Q,K,V in [B,H,S,dh] fp32; mask [B,S] bool (1 byte).
// Block = 128 thr (4 waves).  Q tile 64 rows (16 per wave), KV tiles of 64.
// Q pre-scaled by (1/sqrt(dh))*log2(e) so softmax uses exp2.
// Output merged-head [B,S,D] into workspace for the Wo projection.
// ---------------------------------------------------------------------------
__global__ __launch_bounds__(128) void attn_flash(const float* __restrict__ Q,
                                                  const float* __restrict__ K,
                                                  const float* __restrict__ V,
                                                  const unsigned char* __restrict__ mask,
                                                  float* __restrict__ O) {
  __shared__ unsigned short sQ[64 * 72];
  __shared__ unsigned short sK[64 * 72];
  __shared__ unsigned short sVt[64 * 72];
  __shared__ unsigned short sP[4 * 16 * 72];

  const int tid  = threadIdx.x;
  const int lane = tid & 31;
  const int w    = tid >> 5;      // wave 0..3
  const int lg   = lane >> 4;
  const int ln   = lane & 15;

  const int t  = blockIdx.x;
  const int qt = t & 31;
  const int h  = (t >> 5) & 15;
  const int b  = t >> 9;

  const float QSCALE = 0.125f * 1.44269504088896f;  // 1/sqrt(64) * log2(e)

  const size_t headBase = (size_t)(b * HH + h) * SS * DH;

  // stage Q tile (scaled, bf16)
  {
    const int r0 = tid >> 4;            // 0..7
    const int c4 = (tid & 15) * 4;      // 0..60
#pragma unroll
    for (int i = 0; i < 8; ++i) {
      const int r = r0 + i * 8;
      float4 qv = *reinterpret_cast<const float4*>(Q + headBase + (size_t)(qt * 64 + r) * DH + c4);
      sQ[r * 72 + c4 + 0] = f2bf(qv.x * QSCALE);
      sQ[r * 72 + c4 + 1] = f2bf(qv.y * QSCALE);
      sQ[r * 72 + c4 + 2] = f2bf(qv.z * QSCALE);
      sQ[r * 72 + c4 + 3] = f2bf(qv.w * QSCALE);
    }
  }
  __syncthreads();

  v16bf qa0 = ld_frag(sQ, 72, w * 16, 0);
  v16bf qa1 = ld_frag(sQ, 72, w * 16, 32);

  v8f Oacc[4];
#pragma unroll
  for (int i = 0; i < 4; ++i) Oacc[i] = v8f{0, 0, 0, 0, 0, 0, 0, 0};
  float mi_[8], li_[8];
#pragma unroll
  for (int v = 0; v < 8; ++v) { mi_[v] = -1e30f; li_[v] = 0.0f; }

  for (int kt = 0; kt < 32; ++kt) {
    // stage K tile and V tile (V transposed) as bf16
    {
      const int r0 = tid >> 4;
      const int c4 = (tid & 15) * 4;
#pragma unroll
      for (int i = 0; i < 8; ++i) {
        const int r = r0 + i * 8;
        const size_t src = headBase + (size_t)(kt * 64 + r) * DH + c4;
        float4 kv = *reinterpret_cast<const float4*>(K + src);
        float4 vv = *reinterpret_cast<const float4*>(V + src);
        sK[r * 72 + c4 + 0] = f2bf(kv.x);
        sK[r * 72 + c4 + 1] = f2bf(kv.y);
        sK[r * 72 + c4 + 2] = f2bf(kv.z);
        sK[r * 72 + c4 + 3] = f2bf(kv.w);
        sVt[(c4 + 0) * 72 + r] = f2bf(vv.x);
        sVt[(c4 + 1) * 72 + r] = f2bf(vv.y);
        sVt[(c4 + 2) * 72 + r] = f2bf(vv.z);
        sVt[(c4 + 3) * 72 + r] = f2bf(vv.w);
      }
    }
    __syncthreads();

    // scores S = Q * K^T  (16 q-rows x 64 keys, already in exp2 domain)
    v8f Sc[4];
#pragma unroll
    for (int nf = 0; nf < 4; ++nf) {
      v8f z = v8f{0, 0, 0, 0, 0, 0, 0, 0};
      z = wmma_bf16(qa0, ld_frag(sK, 72, nf * 16, 0), z);
      Sc[nf] = wmma_bf16(qa1, ld_frag(sK, 72, nf * 16, 32), z);
    }

    // padding mask: True => -inf-like
#pragma unroll
    for (int nf = 0; nf < 4; ++nf) {
      const int key = kt * 64 + nf * 16 + ln;
      if (mask[b * SS + key]) {
#pragma unroll
        for (int v = 0; v < 8; ++v) Sc[nf][v] = -1e9f;
      }
    }

    // online softmax update (row lives across 16 lanes of a half-wave)
#pragma unroll
    for (int v = 0; v < 8; ++v) {
      float tmax = fmaxf(fmaxf(Sc[0][v], Sc[1][v]), fmaxf(Sc[2][v], Sc[3][v]));
      tmax = fmaxf(tmax, __shfl_xor(tmax, 1));
      tmax = fmaxf(tmax, __shfl_xor(tmax, 2));
      tmax = fmaxf(tmax, __shfl_xor(tmax, 4));
      tmax = fmaxf(tmax, __shfl_xor(tmax, 8));
      const float mnew  = fmaxf(mi_[v], tmax);
      const float alpha = exp2f(mi_[v] - mnew);
      mi_[v] = mnew;
      float rs = 0.0f;
#pragma unroll
      for (int nf = 0; nf < 4; ++nf) {
        const float p = exp2f(Sc[nf][v] - mnew);
        Sc[nf][v] = p;
        rs += p;
      }
      rs += __shfl_xor(rs, 1);
      rs += __shfl_xor(rs, 2);
      rs += __shfl_xor(rs, 4);
      rs += __shfl_xor(rs, 8);
      li_[v] = li_[v] * alpha + rs;
#pragma unroll
      for (int nf = 0; nf < 4; ++nf) Oacc[nf][v] *= alpha;
    }

    // write P to LDS to re-layout as an A fragment
    unsigned short* sPw = sP + w * 16 * 72;
#pragma unroll
    for (int nf = 0; nf < 4; ++nf)
#pragma unroll
      for (int v = 0; v < 8; ++v)
        sPw[(v + 8 * lg) * 72 + nf * 16 + ln] = f2bf(Sc[nf][v]);

    // O += P * V   (in-wave LDS RAW, DScnt keeps order)
#pragma unroll
    for (int ks = 0; ks < 2; ++ks) {
      v16bf pa = ld_frag(sPw, 72, 0, ks * 32);
#pragma unroll
      for (int nf = 0; nf < 4; ++nf)
        Oacc[nf] = wmma_bf16(pa, ld_frag(sVt, 72, nf * 16, ks * 32), Oacc[nf]);
    }
    __syncthreads();
  }

  // epilogue: normalize and write merged-head [B,S,D]
  float inv[8];
#pragma unroll
  for (int v = 0; v < 8; ++v) inv[v] = 1.0f / li_[v];
#pragma unroll
  for (int nf = 0; nf < 4; ++nf) {
    const int col = h * DH + nf * 16 + ln;
#pragma unroll
    for (int v = 0; v < 8; ++v) {
      const int s = qt * 64 + w * 16 + v + 8 * lg;
      O[((size_t)(b * SS + s)) * DD + col] = Oacc[nf][v] * inv[v];
    }
  }
}

// ---------------------------------------------------------------------------
extern "C" void kernel_launch(void* const* d_in, const int* in_sizes, int n_in,
                              void* d_out, int out_size, void* d_ws, size_t ws_size,
                              hipStream_t stream) {
  (void)in_sizes; (void)n_in; (void)out_size; (void)ws_size;
  const float* x  = (const float*)d_in[0];
  const float* Wq = (const float*)d_in[1];
  const float* Wk = (const float*)d_in[2];
  const float* Wv = (const float*)d_in[3];
  const float* Wo = (const float*)d_in[4];
  const float* bo = (const float*)d_in[5];
  const unsigned char* mask = (const unsigned char*)d_in[6];  // numpy bool, 1 byte

  const size_t HEADSZ = (size_t)BB * HH * SS * DH;  // == B*S*D = 8388608
  float* out_a = (float*)d_out;                     // [B,S,D]
  float* out_k = out_a + HEADSZ;                    // present[0] = K [B,H,S,dh]
  float* out_v = out_k + HEADSZ;                    // present[1] = V [B,H,S,dh]

  float* q_ws    = (float*)d_ws;                    // Q [B,H,S,dh]
  float* attn_ws = q_ws + HEADSZ;                   // attn out [B,S,D]

  dim3 gG(DD / 128, MM / 128);                      // 8 x 64
  linear_wmma<true, false><<<gG, 256, 0, stream>>>(x, Wq, nullptr, q_ws);
  linear_wmma<true, false><<<gG, 256, 0, stream>>>(x, Wk, nullptr, out_k);
  linear_wmma<true, false><<<gG, 256, 0, stream>>>(x, Wv, nullptr, out_v);

  attn_flash<<<dim3(BB * HH * (SS / 64)), 128, 0, stream>>>(q_ws, out_k, out_v,
                                                            mask, attn_ws);

  linear_wmma<false, true><<<gG, 256, 0, stream>>>(attn_ws, Wo, bo, out_a);
}
